// MultiHeadAttention_6923487281352
// MI455X (gfx1250) — matrix operand order repaired and checked by reading, not compiled, not run on hardware
//
#include <hip/hip_runtime.h>
#include <hip/hip_bf16.h>

// ---------------------------------------------------------------------------
// Multi-head attention for MI455X (gfx1250, wave32, WMMA bf16 w/ f32 accum)
//   x:(4,2048,512) f32; 8 heads, head_dim 64.
//   Pipeline: f32->bf16 convert -> 3x WMMA GEMM (QKV, bias fused, [b,h,n,d]
//   bf16 out, W slab LDS-resident via async-to-LDS, barrier-free K loop) ->
//   flash-attention (online softmax, async K staging) -> WMMA GEMM output
//   projection (f32 out).
// ---------------------------------------------------------------------------

typedef __attribute__((ext_vector_type(16))) __bf16 v16bf;
typedef __attribute__((ext_vector_type(8)))  __bf16 v8bf;
typedef __attribute__((ext_vector_type(8)))  float  v8f;

#define EMB   512
#define HEADS 8
#define HDIM  64
#define NSEQ  2048
#define BATCH 4
#define MROWS (BATCH * NSEQ)   // 8192

__device__ __forceinline__ v8f wmma_bf16(v16bf a, v16bf b, v8f c) {
  // D = A(16x32 bf16) x B(32x16 bf16) + C(16x16 f32)
  return __builtin_amdgcn_wmma_f32_16x16x32_bf16(
      /*neg_a=*/false, a, /*neg_b=*/false, b,
      /*c_mod=*/(short)0, c, /*reuse_a=*/false, /*reuse_b=*/false);
}

__device__ __forceinline__ v16bf cat16(v8bf lo, v8bf hi) {
  v16bf r;
#pragma unroll
  for (int i = 0; i < 8; ++i) { r[i] = lo[i]; r[i + 8] = hi[i]; }
  return r;
}

// Low 32 bits of a generic pointer to LDS == LDS byte address (aperture keeps
// the offset in addr[31:0]).
__device__ __forceinline__ unsigned lds_addr32(const void* p) {
  return (unsigned)(unsigned long long)p;
}

// GLOBAL_LOAD_ASYNC_TO_LDS_B128: memory -> LDS, tracked by ASYNCcnt; no VGPR
// data path. (inline asm: portable across ROCm7.2 / amdgpu-toolchain.)
__device__ __forceinline__ void async_ld_b128(unsigned lds, const void* gaddr) {
  asm volatile("global_load_async_to_lds_b128 %0, %1, off"
               :: "v"(lds), "v"((unsigned long long)gaddr) : "memory");
}
__device__ __forceinline__ void wait_asynccnt0() {
  asm volatile("s_wait_asynccnt 0" ::: "memory");
}

// ---------------------------------------------------------------------------
// f32 -> bf16 elementwise convert
// ---------------------------------------------------------------------------
__global__ void cvt_f32_bf16(const float* __restrict__ in,
                             __bf16* __restrict__ out, int n) {
  int i = blockIdx.x * blockDim.x + threadIdx.x;
  if (i < n) out[i] = (__bf16)in[i];
}

// ---------------------------------------------------------------------------
// C[m, j] = sum_k A[m,k] * W[j,k] + bias[j]
//   A: [8192, 512] bf16 row-major, W: [512, 512] bf16 row-major, bias f32.
//   layout 0: out bf16 scattered into [b, h, n, d]  (for Q/K/V)
//   layout 1: out f32 row-major [8192, 512]         (final projection)
// Workgroup: 128 threads (4 waves), tile 128(M) x 64(N).
// The 64x512 W slab (contiguous, 64 KB) is async-staged into LDS ONCE; the
// whole K loop then runs barrier-free: A fragments straight from global
// (L2-resident), B fragments from LDS, 8 WMMAs per 32-K step.
// ---------------------------------------------------------------------------
__global__ __launch_bounds__(128)
void gemm_bias_wmma(const __bf16* __restrict__ A, const __bf16* __restrict__ W,
                    const float* __restrict__ bias, void* __restrict__ outp,
                    int layout) {
  __shared__ __bf16 Wl[64 * EMB];     // 64 KB W slab: rows n0..n0+63, all K

  const int m0   = blockIdx.x * 128;
  const int n0   = blockIdx.y * 64;
  const int tid  = threadIdx.x;
  const int wave = tid >> 5;
  const int lane = tid & 31;
  const int l16  = lane & 15;
  const int hi   = lane >> 4;

  // ---- one-shot async stage of the contiguous 64x512 W slab ----
  {
    const __bf16* wsrc = W + (size_t)n0 * EMB;    // contiguous 32768 halfs
    const unsigned lbase = lds_addr32(Wl);
#pragma unroll 4
    for (int i = 0; i < 32; ++i) {                // 4096 x 16-half chunks
      const int off = (i * 128 + tid) * 16;       // halfs
      async_ld_b128(lbase + off * 2, wsrc + off);
    }
    wait_asynccnt0();
  }
  __syncthreads();

  // ---- barrier-free K loop: each wave owns a 32(M) x 64(N) output tile ----
  const int row0 = m0 + wave * 32;
  v8f acc[2][4] = {};

  for (int k0 = 0; k0 < EMB; k0 += 32) {
    // A fragments (16x32 each): lane<16 -> row=l16, K={0..7,16..23};
    // lane>=16 -> K={8..15,24..31}
    v16bf af[2];
#pragma unroll
    for (int rg = 0; rg < 2; ++rg) {
      const __bf16* p = A + (size_t)(row0 + rg * 16 + l16) * EMB + k0 + hi * 8;
      af[rg] = cat16(*(const v8bf*)p, *(const v8bf*)(p + 16));
    }
#pragma unroll
    for (int nc = 0; nc < 4; ++nc) {
      // B fragment: 32x16, col n = nc*16 + l16, K={hi*8..+7, 16+hi*8..+7}
      const __bf16* p = Wl + (nc * 16 + l16) * EMB + k0 + hi * 8;
      v16bf bf_ = cat16(*(const v8bf*)p, *(const v8bf*)(p + 16));
      acc[0][nc] = wmma_bf16(af[0], bf_, acc[0][nc]);
      acc[1][nc] = wmma_bf16(af[1], bf_, acc[1][nc]);
    }
  }

  // ---- epilogue: bias + store (D layout: row M = v + 8*hi, col N = l16) ----
#pragma unroll
  for (int rg = 0; rg < 2; ++rg) {
    const int mrow0 = row0 + rg * 16;
#pragma unroll
    for (int nc = 0; nc < 4; ++nc) {
      const int col = n0 + nc * 16 + l16;
      const float bv = bias[col];
      if (layout == 1) {
        float* out = (float*)outp;
#pragma unroll
        for (int v = 0; v < 8; ++v)
          out[(size_t)(mrow0 + v + hi * 8) * EMB + col] = acc[rg][nc][v] + bv;
      } else {
        __bf16* out = (__bf16*)outp;
#pragma unroll
        for (int v = 0; v < 8; ++v) {
          const int m  = mrow0 + v + hi * 8;
          const int b  = m >> 11;           // /2048
          const int n  = m & 2047;
          const int hh = col >> 6;          // /64
          const int dd = col & 63;
          out[(((size_t)(b * HEADS + hh) * NSEQ + n) * HDIM) + dd] =
              (__bf16)(acc[rg][nc][v] + bv);
        }
      }
    }
  }
}

// ---------------------------------------------------------------------------
// Flash attention: grid (b*h = 32, n/64 = 32), 128 threads (4 waves).
// Each wave owns a 16-row q tile; loops over keys in blocks of 32 with
// online softmax. Logits never leave registers. scale = 1/sqrt(EMB).
// K tile staged via async-to-LDS; V staged transposed via VGPR path.
// ---------------------------------------------------------------------------
__global__ __launch_bounds__(128)
void flash_attn_wmma(const __bf16* __restrict__ Q, const __bf16* __restrict__ K,
                     const __bf16* __restrict__ V, __bf16* __restrict__ O) {
  __shared__ __bf16 Kl[32 * 64];       // 32 keys x 64 dims, row-major (async)
  __shared__ __bf16 Vt[64 * 32];       // transposed: [dim][key]
  __shared__ __bf16 Pl[4 * 16 * 32];   // per-wave P staging (D->A relayout)

  const int bh   = blockIdx.x;         // b*8 + h
  const int qb   = blockIdx.y;         // q block of 64 rows
  const int tid  = threadIdx.x;
  const int wave = tid >> 5;
  const int lane = tid & 31;
  const int l16  = lane & 15;
  const int hi   = lane >> 4;

  const __bf16* Qb = Q + (size_t)bh * NSEQ * HDIM;
  const __bf16* Kb = K + (size_t)bh * NSEQ * HDIM;
  const __bf16* Vb = V + (size_t)bh * NSEQ * HDIM;
  const unsigned klbase = lds_addr32(Kl);

  // Q fragments for this wave's 16 rows: two 16x32 A tiles (d=0..31, 32..63)
  v16bf aq0, aq1;
  {
    const int qrow = qb * 64 + wave * 16 + l16;
    const __bf16* qp = Qb + (size_t)qrow * HDIM;
    aq0 = cat16(*(const v8bf*)(qp + hi * 8),      *(const v8bf*)(qp + hi * 8 + 16));
    aq1 = cat16(*(const v8bf*)(qp + 32 + hi * 8), *(const v8bf*)(qp + 32 + hi * 8 + 16));
  }

  v8f acc[4] = {};
  float mrow[8], lrow[8];
#pragma unroll
  for (int v = 0; v < 8; ++v) { mrow[v] = -1e30f; lrow[v] = 0.0f; }

  const float scale = 0.04419417382415922f;  // 1/sqrt(512)
  __bf16* Pw = Pl + wave * 16 * 32;

  for (int kb = 0; kb < NSEQ; kb += 32) {
    __syncthreads();
    // K block (32x64 halfs, contiguous): async memory -> LDS, 1 chunk/thread
    async_ld_b128(klbase + tid * 32, Kb + (size_t)kb * HDIM + tid * 16);
    // V block: global -> VGPR -> transposed LDS store
    {
      const int row = tid >> 2;            // key 0..31
      const int cb  = (tid & 3) * 16;      // dim base 0/16/32/48
      const __bf16* sv = Vb + (size_t)(kb + row) * HDIM + cb;
      v8bf v0 = *(const v8bf*)(sv), v1 = *(const v8bf*)(sv + 8);
#pragma unroll
      for (int i = 0; i < 8; ++i) {
        Vt[(cb + i) * 32 + row]     = v0[i];
        Vt[(cb + 8 + i) * 32 + row] = v1[i];
      }
    }
    if (kb + 32 < NSEQ) __builtin_prefetch(Vb + (size_t)(kb + 32) * HDIM, 0, 1);
    wait_asynccnt0();
    __syncthreads();

    // S = Q Kt for two 16-key tiles (K-dim = head dim, chunks 0..31 / 32..63)
    auto loadBK = [&](int t, int d0) -> v16bf {
      const __bf16* p = Kl + (t * 16 + l16) * 64 + d0 + hi * 8;
      return cat16(*(const v8bf*)p, *(const v8bf*)(p + 16));
    };
    v8f s0 = {}, s1 = {};
    s0 = wmma_bf16(aq0, loadBK(0, 0),  s0);
    s0 = wmma_bf16(aq1, loadBK(0, 32), s0);
    s1 = wmma_bf16(aq0, loadBK(1, 0),  s1);
    s1 = wmma_bf16(aq1, loadBK(1, 32), s1);

    // online softmax over the 32 keys of this block, row stats per VGPR slot
#pragma unroll
    for (int v = 0; v < 8; ++v) {
      const float e0 = s0[v] * scale;
      const float e1 = s1[v] * scale;
      float c = fmaxf(e0, e1);
#pragma unroll
      for (int off = 8; off > 0; off >>= 1) c = fmaxf(c, __shfl_xor(c, off, 32));
      const float mn = fmaxf(mrow[v], c);
      const float p0 = __expf(e0 - mn);
      const float p1 = __expf(e1 - mn);
      float r = p0 + p1;
#pragma unroll
      for (int off = 8; off > 0; off >>= 1) r += __shfl_xor(r, off, 32);
      const float f = __expf(mrow[v] - mn);
      lrow[v] = lrow[v] * f + r;
      mrow[v] = mn;
      acc[0][v] *= f; acc[1][v] *= f; acc[2][v] *= f; acc[3][v] *= f;
      // stage P in row-major [row][key] for A-fragment reload
      const int prow = v + hi * 8;
      Pw[prow * 32 + l16]      = (__bf16)p0;
      Pw[prow * 32 + 16 + l16] = (__bf16)p1;
    }
    asm volatile("s_wait_dscnt 0" ::: "memory");  // P writes -> P reads (same wave)

    // A fragment of P (16 rows x 32 keys)
    v16bf ap;
    {
      const __bf16* p = Pw + l16 * 32 + hi * 8;
      ap = cat16(*(const v8bf*)p, *(const v8bf*)(p + 16));
    }
    // acc += P x V  (V as 32x16 B fragments from transposed LDS tile)
#pragma unroll
    for (int nc = 0; nc < 4; ++nc) {
      const __bf16* p = Vt + (nc * 16 + l16) * 32 + hi * 8;
      v16bf bv = cat16(*(const v8bf*)p, *(const v8bf*)(p + 16));
      acc[nc] = wmma_bf16(ap, bv, acc[nc]);
    }
  }

  // epilogue: normalize and write O as [m = b*2048+n, col = h*64+d] bf16
  const int h = bh & 7;
  const int b = bh >> 3;
  const size_t mbase = (size_t)b * NSEQ + qb * 64 + wave * 16;
  float invl[8];
#pragma unroll
  for (int v = 0; v < 8; ++v) invl[v] = 1.0f / lrow[v];
#pragma unroll
  for (int nc = 0; nc < 4; ++nc) {
    const int col = h * HDIM + nc * 16 + l16;
#pragma unroll
    for (int v = 0; v < 8; ++v)
      O[(mbase + v + hi * 8) * EMB + col] = (__bf16)(acc[nc][v] * invl[v]);
  }
}

// ---------------------------------------------------------------------------
extern "C" void kernel_launch(void* const* d_in, const int* in_sizes, int n_in,
                              void* d_out, int out_size, void* d_ws, size_t ws_size,
                              hipStream_t stream) {
  const float* x  = (const float*)d_in[0];
  const float* Wq = (const float*)d_in[1];
  const float* bq = (const float*)d_in[2];
  const float* Wk = (const float*)d_in[3];
  const float* bk = (const float*)d_in[4];
  const float* Wv = (const float*)d_in[5];
  const float* bv = (const float*)d_in[6];
  const float* Wo = (const float*)d_in[7];
  const float* bo = (const float*)d_in[8];

  char* ws = (char*)d_ws;
  const size_t MB = 1024ull * 1024ull;
  __bf16* xb  = (__bf16*)(ws + 0);          //  8 MiB: x bf16 [8192,512]
  __bf16* wqb = (__bf16*)(ws + 8  * MB);    //  0.5 MiB each
  __bf16* wkb = (__bf16*)(ws + 8  * MB + 512 * 1024);
  __bf16* wvb = (__bf16*)(ws + 9  * MB);
  __bf16* wob = (__bf16*)(ws + 9  * MB + 512 * 1024);
  __bf16* Qb  = (__bf16*)(ws + 10 * MB);    //  8 MiB [b,h,n,d]
  __bf16* Kb  = (__bf16*)(ws + 18 * MB);
  __bf16* Vb  = (__bf16*)(ws + 26 * MB);
  __bf16* Ob  = (__bf16*)(ws + 34 * MB);    //  8 MiB [8192,512]

  // 1) convert inputs to bf16
  {
    int n = MROWS * EMB;
    cvt_f32_bf16<<<(n + 255) / 256, 256, 0, stream>>>(x, xb, n);
    n = EMB * EMB;
    cvt_f32_bf16<<<(n + 255) / 256, 256, 0, stream>>>(Wq, wqb, n);
    cvt_f32_bf16<<<(n + 255) / 256, 256, 0, stream>>>(Wk, wkb, n);
    cvt_f32_bf16<<<(n + 255) / 256, 256, 0, stream>>>(Wv, wvb, n);
    cvt_f32_bf16<<<(n + 255) / 256, 256, 0, stream>>>(Wo, wob, n);
  }

  // 2) QKV projections (bias fused, bf16 [b,h,n,d] outputs)
  dim3 ggrid(MROWS / 128, EMB / 64);
  gemm_bias_wmma<<<ggrid, 128, 0, stream>>>(xb, wqb, bq, (void*)Qb, 0);
  gemm_bias_wmma<<<ggrid, 128, 0, stream>>>(xb, wkb, bk, (void*)Kb, 0);
  gemm_bias_wmma<<<ggrid, 128, 0, stream>>>(xb, wvb, bv, (void*)Vb, 0);

  // 3) flash attention
  flash_attn_wmma<<<dim3(BATCH * HEADS, NSEQ / 64), 128, 0, stream>>>(Qb, Kb, Vb, Ob);

  // 4) output projection -> f32 d_out
  gemm_bias_wmma<<<ggrid, 128, 0, stream>>>(Ob, wob, bo, d_out, 1);
}
